// PatchedAttention_12086037971114
// MI455X (gfx1250) — compile-verified, matrix operand order checked by reading, not run on hardware
//
#include <hip/hip_runtime.h>
#include <hip/hip_bf16.h>
#include <stdint.h>

// ---------------------------------------------------------------------------
// Patched attention for MI455X (gfx1250), wave32 + WMMA bf16 + async-to-LDS.
// ---------------------------------------------------------------------------

#define DMODEL 1024
#define NHEADS 16
#define DHEAD  64
#define PATCHN 128
#define MTOT   32768   // B*T = 4*8192

typedef __attribute__((ext_vector_type(16))) __bf16 v16bf;
typedef __attribute__((ext_vector_type(8)))  float  v8f;

union Frag { v16bf v; uint4 u[2]; };

// ---- fp32 -> bf16 -------------------------------------------------------
static __device__ __forceinline__ uint16_t f2bf(float f) {
  union { float f; uint32_t u; } x; x.f = f;
  uint32_t r = x.u + 0x7FFFu + ((x.u >> 16) & 1u);   // round-to-nearest-even
  return (uint16_t)(r >> 16);
}
static __device__ __forceinline__ uint32_t pack2(float lo, float hi) {
#if __has_builtin(__builtin_amdgcn_cvt_pk_bf16_f32)
  auto p = __builtin_amdgcn_cvt_pk_bf16_f32(lo, hi);
  uint32_t u; __builtin_memcpy(&u, &p, sizeof(u));
  return u;
#else
  return (uint32_t)f2bf(lo) | ((uint32_t)f2bf(hi) << 16);
#endif
}

// ---- async global -> LDS (gfx1250), sync fallback -----------------------
#if __has_builtin(__builtin_amdgcn_global_load_async_to_lds_b128) && \
    __has_builtin(__builtin_amdgcn_s_wait_asynccnt)
#define HAS_ASYNC 1
typedef int v4i_ __attribute__((vector_size(4 * sizeof(int))));
typedef __attribute__((address_space(1))) v4i_ gv4i;
typedef __attribute__((address_space(3))) v4i_ lv4i;
#else
#define HAS_ASYNC 0
#endif

static __device__ __forceinline__ void cp16_async(void* lds, const void* g) {
#if HAS_ASYNC
  __builtin_amdgcn_global_load_async_to_lds_b128((gv4i*)g, (lv4i*)lds, 0, 0);
#else
  *(uint4*)lds = *(const uint4*)g;
#endif
}
static __device__ __forceinline__ void async_wait() {
#if HAS_ASYNC
  __builtin_amdgcn_s_wait_asynccnt(0);
#endif
}

// ---------------------------------------------------------------------------
// prep_x: x fp32 -> bf16, one pass (8 elements / thread)
// ---------------------------------------------------------------------------
__global__ void __launch_bounds__(256)
prep_x(const float* __restrict__ x, uint16_t* __restrict__ Xb) {
  const size_t i = ((size_t)blockIdx.x * 256 + threadIdx.x) * 8;
  float4 f0 = ((const float4*)(x + i))[0];
  float4 f1 = ((const float4*)(x + i))[1];
  uint4 o = make_uint4(pack2(f0.x, f0.y), pack2(f0.z, f0.w),
                       pack2(f1.x, f1.y), pack2(f1.z, f1.w));
  *(uint4*)(Xb + i) = o;
}

// ---------------------------------------------------------------------------
// prep_w: Wt[n][k] = bf16(W[k][n])   (32x32 LDS transpose)
// ---------------------------------------------------------------------------
__global__ void __launch_bounds__(256)
prep_w(const float* __restrict__ W, uint16_t* __restrict__ Wt) {
  __shared__ float tile[32][33];
  const int bx = blockIdx.x & 31, by = blockIdx.x >> 5;
  const int tx = threadIdx.x & 31, ty = threadIdx.x >> 5;
  #pragma unroll
  for (int r = ty; r < 32; r += 8)
    tile[r][tx] = W[(size_t)(by * 32 + r) * DMODEL + bx * 32 + tx];
  __syncthreads();
  #pragma unroll
  for (int r = ty; r < 32; r += 8)
    Wt[(size_t)(bx * 32 + r) * DMODEL + by * 32 + tx] = f2bf(tile[tx][r]);
}

// ---------------------------------------------------------------------------
// QKV GEMM: Xb(bf16 [M][1024]) @ Wt^T + b.
// Block tile 128(M) x 128(N), 8 waves (2x4), wave tile 64x32 (4x2 WMMA,
// acc = 64 VGPRs so the allocator can tie WMMA D=C in place).
// Double-buffered async staging. sel: 0=Q row-major, 1=K row-major,
// 2=V transposed per (patch,head).
// ---------------------------------------------------------------------------
__global__ void __launch_bounds__(256)
gemm_qkv(const uint16_t* __restrict__ Xb,
         const uint16_t* __restrict__ Wqt, const uint16_t* __restrict__ Wkt,
         const uint16_t* __restrict__ Wvt,
         const float* __restrict__ bq, const float* __restrict__ bk,
         const float* __restrict__ bv,
         uint16_t* __restrict__ Qb, uint16_t* __restrict__ Kb,
         uint16_t* __restrict__ Vt) {
  const int m0  = blockIdx.x * 128;
  const int n0  = blockIdx.y * 128;
  const int sel = blockIdx.z;
  const uint16_t* Bt  = (sel == 0) ? Wqt : (sel == 1) ? Wkt : Wvt;
  const float*   bias = (sel == 0) ? bq  : (sel == 1) ? bk  : bv;

  __shared__ __align__(16) uint16_t As[2][128 * 32];
  __shared__ __align__(16) uint16_t Bs[2][128 * 32];

  const int tid  = threadIdx.x;
  const int wave = tid >> 5, lane = tid & 31;
  const int wm = wave >> 2, wn = wave & 3;     // 2 x 4 wave grid
  const int lrow = lane & 15;
  const int kh   = (lane >> 4) * 8;

  auto stage = [&](int buf, int k0) {
    const int row = tid >> 1, c = (tid & 1) * 16;
    {
      const uint16_t* g = Xb + (size_t)(m0 + row) * DMODEL + k0 + c;
      cp16_async(&As[buf][row * 32 + c], g);
      cp16_async(&As[buf][row * 32 + c + 8], g + 8);
    }
    {
      const uint16_t* g = Bt + (size_t)(n0 + row) * DMODEL + k0 + c;
      cp16_async(&Bs[buf][row * 32 + c], g);
      cp16_async(&Bs[buf][row * 32 + c + 8], g + 8);
    }
  };

  v8f acc[4][2] = {};
  stage(0, 0);
  for (int ks = 0; ks < DMODEL / 32; ++ks) {
    async_wait();
    __syncthreads();
    if (ks + 1 < DMODEL / 32) stage((ks + 1) & 1, (ks + 1) * 32);

    const uint16_t* Ab = As[ks & 1];
    const uint16_t* Bb = Bs[ks & 1];
    Frag b[2];
    #pragma unroll
    for (int j = 0; j < 2; ++j) {
      const uint4* p = (const uint4*)&Bb[(wn * 32 + j * 16 + lrow) * 32 + kh];
      b[j].u[0] = p[0]; b[j].u[1] = p[2];
    }
    #pragma unroll
    for (int i = 0; i < 4; ++i) {
      Frag a;
      const uint4* p = (const uint4*)&Ab[(wm * 64 + i * 16 + lrow) * 32 + kh];
      a.u[0] = p[0]; a.u[1] = p[2];
      #pragma unroll
      for (int j = 0; j < 2; ++j)
        acc[i][j] = __builtin_amdgcn_wmma_f32_16x16x32_bf16(
            false, a.v, false, b[j].v, (short)0, acc[i][j], false, false);
    }
  }

  if (sel < 2) {
    uint16_t* Out = (sel == 0) ? Qb : Kb;
    #pragma unroll
    for (int i = 0; i < 4; ++i) {
      const int mrow = m0 + wm * 64 + i * 16 + kh;
      #pragma unroll
      for (int j = 0; j < 2; ++j) {
        const int n = n0 + wn * 32 + j * 16 + lrow;
        const float bb = bias[n];
        #pragma unroll
        for (int r = 0; r < 8; ++r)
          Out[(size_t)(mrow + r) * DMODEL + n] = f2bf(acc[i][j][r] + bb);
      }
    }
  } else {
    #pragma unroll
    for (int i = 0; i < 4; ++i) {
      const int gm = m0 + wm * 64 + i * 16 + kh;
      const int patch = gm >> 7, tok = gm & 127;
      #pragma unroll
      for (int j = 0; j < 2; ++j) {
        const int n = n0 + wn * 32 + j * 16 + lrow;
        const int head = n >> 6, nl = n & 63;
        const float bb = bias[n];
        uint4 pk = make_uint4(pack2(acc[i][j][0] + bb, acc[i][j][1] + bb),
                              pack2(acc[i][j][2] + bb, acc[i][j][3] + bb),
                              pack2(acc[i][j][4] + bb, acc[i][j][5] + bb),
                              pack2(acc[i][j][6] + bb, acc[i][j][7] + bb));
        *(uint4*)&Vt[((size_t)(patch * NHEADS + head) * DHEAD + nl) * PATCHN + tok] = pk;
      }
    }
  }
}

// ---------------------------------------------------------------------------
// Attention per (patch, head): S = QK^T/8, softmax, O = W V.
// K and V^T tiles staged once into LDS via async copies.
// ---------------------------------------------------------------------------
__global__ void __launch_bounds__(256)
attn_kernel(const uint16_t* __restrict__ Q, const uint16_t* __restrict__ K,
            const uint16_t* __restrict__ Vt, uint16_t* __restrict__ Aout) {
  const int patch = blockIdx.x >> 4;
  const int head  = blockIdx.x & 15;

  const uint16_t* Qp  = Q  + (size_t)patch * PATCHN * DMODEL + head * DHEAD;
  const uint16_t* Kp  = K  + (size_t)patch * PATCHN * DMODEL + head * DHEAD;
  const uint16_t* Vtp = Vt + (size_t)(patch * NHEADS + head) * DHEAD * PATCHN;
  uint16_t*       Op  = Aout + (size_t)patch * PATCHN * DMODEL + head * DHEAD;

  __shared__ __align__(16) uint16_t Ks[PATCHN * DHEAD];    // 16 KB, [tok][d]
  __shared__ __align__(16) uint16_t Vs[DHEAD * PATCHN];    // 16 KB, [d][tok]
  __shared__ __align__(16) uint16_t Wlds[PATCHN * PATCHN]; // 32 KB

  const int tid  = threadIdx.x;
  const int wave = tid >> 5, lane = tid & 31;
  const int lrow = lane & 15;
  const int kh   = (lane >> 4) * 8;

  // ---- stage K (128x64) and V^T (64x128) ----
  {
    const int row = tid >> 1, c0 = (tid & 1) * 32;
    const uint16_t* g = Kp + (size_t)row * DMODEL + c0;
    #pragma unroll
    for (int c = 0; c < 32; c += 8)
      cp16_async(&Ks[row * DHEAD + c0 + c], g + c);
  }
  {
    const int row = tid >> 2, c0 = (tid & 3) * 32;
    const uint16_t* g = Vtp + (size_t)row * PATCHN + c0;
    #pragma unroll
    for (int c = 0; c < 32; c += 8)
      cp16_async(&Vs[row * PATCHN + c0 + c], g + c);
  }
  async_wait();
  __syncthreads();

  // ---- phase 1: S = Q K^T ----
  v8f acc[8] = {};
  #pragma unroll
  for (int ks = 0; ks < DHEAD; ks += 32) {
    Frag a;
    {
      const uint4* p = (const uint4*)(Qp + (size_t)(wave * 16 + lrow) * DMODEL + ks + kh);
      a.u[0] = p[0]; a.u[1] = p[2];
    }
    #pragma unroll
    for (int t = 0; t < 8; ++t) {
      Frag b;
      const uint4* p = (const uint4*)&Ks[(t * 16 + lrow) * DHEAD + ks + kh];
      b.u[0] = p[0]; b.u[1] = p[2];
      acc[t] = __builtin_amdgcn_wmma_f32_16x16x32_bf16(
          false, a.v, false, b.v, (short)0, acc[t], false, false);
    }
  }

  // ---- phase 2: softmax rows + transpose to LDS as bf16 ----
  const float C = 0.125f * 1.44269504088896f;  // (1/sqrt(64)) * log2(e)
  #pragma unroll
  for (int r = 0; r < 8; ++r) {
    float mv = acc[0][r];
    #pragma unroll
    for (int t = 1; t < 8; ++t) mv = fmaxf(mv, acc[t][r]);
    #pragma unroll
    for (int off = 1; off <= 8; off <<= 1)
      mv = fmaxf(mv, __shfl_xor(mv, off, 32));   // 16-lane half reduce
    float sum = 0.f;
    #pragma unroll
    for (int t = 0; t < 8; ++t) {
      float e = exp2f((acc[t][r] - mv) * C);
      acc[t][r] = e;
      sum += e;
    }
    #pragma unroll
    for (int off = 1; off <= 8; off <<= 1)
      sum += __shfl_xor(sum, off, 32);
    const float inv = 1.0f / sum;
    const int row = wave * 16 + kh + r;
    #pragma unroll
    for (int t = 0; t < 8; ++t)
      Wlds[row * PATCHN + t * 16 + lrow] = f2bf(acc[t][r] * inv);
  }
  __syncthreads();

  // ---- phase 3: O = W V ----
  v8f oacc[4] = {};
  #pragma unroll
  for (int ksec = 0; ksec < 4; ++ksec) {
    Frag a;
    {
      const uint4* p = (const uint4*)&Wlds[(wave * 16 + lrow) * PATCHN + ksec * 32 + kh];
      a.u[0] = p[0]; a.u[1] = p[2];
    }
    #pragma unroll
    for (int n = 0; n < 4; ++n) {
      Frag b;
      const uint4* p = (const uint4*)&Vs[(n * 16 + lrow) * PATCHN + ksec * 32 + kh];
      b.u[0] = p[0]; b.u[1] = p[2];
      oacc[n] = __builtin_amdgcn_wmma_f32_16x16x32_bf16(
          false, a.v, false, b.v, (short)0, oacc[n], false, false);
    }
  }

  #pragma unroll
  for (int n = 0; n < 4; ++n) {
    const int col = n * 16 + lrow;
    #pragma unroll
    for (int r = 0; r < 8; ++r)
      Op[(size_t)(wave * 16 + kh + r) * DMODEL + col] = f2bf(oacc[n][r]);
  }
}

// ---------------------------------------------------------------------------
// Output GEMM: attn(bf16) @ Wo^T + bo -> fp32 out. Same tiling as gemm_qkv.
// ---------------------------------------------------------------------------
__global__ void __launch_bounds__(256)
gemm_out(const uint16_t* __restrict__ A, const uint16_t* __restrict__ Wot,
         const float* __restrict__ bo, float* __restrict__ Out) {
  const int m0 = blockIdx.x * 128;
  const int n0 = blockIdx.y * 128;

  __shared__ __align__(16) uint16_t As[2][128 * 32];
  __shared__ __align__(16) uint16_t Bs[2][128 * 32];

  const int tid  = threadIdx.x;
  const int wave = tid >> 5, lane = tid & 31;
  const int wm = wave >> 2, wn = wave & 3;
  const int lrow = lane & 15;
  const int kh   = (lane >> 4) * 8;

  auto stage = [&](int buf, int k0) {
    const int row = tid >> 1, c = (tid & 1) * 16;
    {
      const uint16_t* g = A + (size_t)(m0 + row) * DMODEL + k0 + c;
      cp16_async(&As[buf][row * 32 + c], g);
      cp16_async(&As[buf][row * 32 + c + 8], g + 8);
    }
    {
      const uint16_t* g = Wot + (size_t)(n0 + row) * DMODEL + k0 + c;
      cp16_async(&Bs[buf][row * 32 + c], g);
      cp16_async(&Bs[buf][row * 32 + c + 8], g + 8);
    }
  };

  v8f acc[4][2] = {};
  stage(0, 0);
  for (int ks = 0; ks < DMODEL / 32; ++ks) {
    async_wait();
    __syncthreads();
    if (ks + 1 < DMODEL / 32) stage((ks + 1) & 1, (ks + 1) * 32);

    const uint16_t* Ab = As[ks & 1];
    const uint16_t* Bb = Bs[ks & 1];
    Frag b[2];
    #pragma unroll
    for (int j = 0; j < 2; ++j) {
      const uint4* p = (const uint4*)&Bb[(wn * 32 + j * 16 + lrow) * 32 + kh];
      b[j].u[0] = p[0]; b[j].u[1] = p[2];
    }
    #pragma unroll
    for (int i = 0; i < 4; ++i) {
      Frag a;
      const uint4* p = (const uint4*)&Ab[(wm * 64 + i * 16 + lrow) * 32 + kh];
      a.u[0] = p[0]; a.u[1] = p[2];
      #pragma unroll
      for (int j = 0; j < 2; ++j)
        acc[i][j] = __builtin_amdgcn_wmma_f32_16x16x32_bf16(
            false, a.v, false, b[j].v, (short)0, acc[i][j], false, false);
    }
  }

  #pragma unroll
  for (int i = 0; i < 4; ++i) {
    const int mrow = m0 + wm * 64 + i * 16 + kh;
    #pragma unroll
    for (int j = 0; j < 2; ++j) {
      const int n = n0 + wn * 32 + j * 16 + lrow;
      const float bb = bo[n];
      #pragma unroll
      for (int r = 0; r < 8; ++r)
        Out[(size_t)(mrow + r) * DMODEL + n] = acc[i][j][r] + bb;
    }
  }
}

// ---------------------------------------------------------------------------
// Launch
// ---------------------------------------------------------------------------
extern "C" void kernel_launch(void* const* d_in, const int* in_sizes, int n_in,
                              void* d_out, int out_size, void* d_ws, size_t ws_size,
                              hipStream_t stream) {
  (void)in_sizes; (void)n_in; (void)out_size; (void)ws_size;

  const float* x  = (const float*)d_in[0];
  // d_in[1] = coords (unused by reference math)
  const float* Wq = (const float*)d_in[2];
  const float* bq = (const float*)d_in[3];
  const float* Wk = (const float*)d_in[4];
  const float* bk = (const float*)d_in[5];
  const float* Wv = (const float*)d_in[6];
  const float* bv = (const float*)d_in[7];
  const float* Wo = (const float*)d_in[8];
  const float* bo = (const float*)d_in[9];
  float* out = (float*)d_out;

  char* ws = (char*)d_ws;
  const size_t WSZ = (size_t)DMODEL * DMODEL * sizeof(uint16_t);   // 2 MiB
  const size_t TSZ = (size_t)MTOT * DMODEL * sizeof(uint16_t);     // 64 MiB
  uint16_t* Wqt  = (uint16_t*)(ws);
  uint16_t* Wkt  = (uint16_t*)(ws + WSZ);
  uint16_t* Wvt  = (uint16_t*)(ws + 2 * WSZ);
  uint16_t* Wot  = (uint16_t*)(ws + 3 * WSZ);
  uint16_t* Xb   = (uint16_t*)(ws + 4 * WSZ);
  uint16_t* Qb   = (uint16_t*)(ws + 4 * WSZ + TSZ);
  uint16_t* Kb   = (uint16_t*)(ws + 4 * WSZ + 2 * TSZ);
  uint16_t* Vt   = (uint16_t*)(ws + 4 * WSZ + 3 * TSZ);
  uint16_t* Attn = (uint16_t*)(ws + 4 * WSZ + 4 * TSZ);

  // 1) one-pass conversions
  prep_x<<<dim3((MTOT * DMODEL) / (256 * 8)), dim3(256), 0, stream>>>(x, Xb);
  prep_w<<<dim3(1024), dim3(256), 0, stream>>>(Wq, Wqt);
  prep_w<<<dim3(1024), dim3(256), 0, stream>>>(Wk, Wkt);
  prep_w<<<dim3(1024), dim3(256), 0, stream>>>(Wv, Wvt);
  prep_w<<<dim3(1024), dim3(256), 0, stream>>>(Wo, Wot);

  // 2) fused QKV projection
  gemm_qkv<<<dim3(MTOT / 128, DMODEL / 128, 3), dim3(256), 0, stream>>>(
      Xb, Wqt, Wkt, Wvt, bq, bk, bv, Qb, Kb, Vt);

  // 3) per-(patch, head) attention
  attn_kernel<<<dim3((MTOT / PATCHN) * NHEADS), dim3(256), 0, stream>>>(Qb, Kb, Vt, Attn);

  // 4) output projection (fp32 + bias)
  gemm_out<<<dim3(MTOT / 128, DMODEL / 128), dim3(256), 0, stream>>>(Attn, Wot, bo, out);
}
